// PointNetSetAbstractionMsg_16234976379619
// MI455X (gfx1250) — compile-verified, matrix operand order checked by reading, not run on hardware
//
#include <hip/hip_runtime.h>
#include <hip/hip_fp16.h>

typedef _Float16 v8h  __attribute__((ext_vector_type(8)));
typedef _Float16 v16h __attribute__((ext_vector_type(16)));
typedef float    v8f  __attribute__((ext_vector_type(8)));

#define NPTS   4096
#define NBATCH 8
#define NPOINT 512
#define OUT_XYZ (NBATCH*3*NPOINT)   /* 12288 floats */
#define TOTCH  320
#define EPSV   1e-5f

#define MODE_STORE 0
#define MODE_STATS 1
#define MODE_FINAL 2

// ---------------------------------------------------------------- zero
__global__ __launch_bounds__(256) void zero_kernel(float* __restrict__ p, long long n) {
  long long i = (long long)blockIdx.x * 256 + threadIdx.x;
  if (i < n) p[i] = 0.f;
}

// ---------------------------------------------------------------- FPS
__global__ __launch_bounds__(512) void fps_kernel(const float* __restrict__ points,
                                                  const int* __restrict__ start,
                                                  float* __restrict__ nxyz,
                                                  float* __restrict__ out) {
  __shared__ float sd[NPTS];
  __shared__ float smax[512];
  __shared__ int   simax[512];
  __shared__ int   sfid[NPOINT];

  const int b = blockIdx.x, tid = threadIdx.x;
  const float* px = points + (size_t)b * 3 * NPTS;

  for (int j = tid; j < NPTS; j += 512) sd[j] = 1e10f;
  __syncthreads();

  int far = start[b];
  for (int t = 0; t < NPOINT; ++t) {
    if (tid == 0) sfid[t] = far;                 // emit BEFORE update (matches lax.scan)
    float cx = px[far], cy = px[NPTS + far], cz = px[2 * NPTS + far];
    float bm = -1.f; int bi = 0x7fffffff;
    for (int j = tid; j < NPTS; j += 512) {
      float dx = px[j] - cx, dy = px[NPTS + j] - cy, dz = px[2 * NPTS + j] - cz;
      float d = dx * dx + dy * dy + dz * dz;
      float o = sd[j];
      d = d < o ? d : o;
      sd[j] = d;
      if (d > bm) { bm = d; bi = j; }            // first-index tie-break (ascending j)
    }
    smax[tid] = bm; simax[tid] = bi;
    __syncthreads();
    for (int off = 256; off; off >>= 1) {
      if (tid < off) {
        float v = smax[tid + off]; int i2 = simax[tid + off];
        if (v > smax[tid] || (v == smax[tid] && i2 < simax[tid])) { smax[tid] = v; simax[tid] = i2; }
      }
      __syncthreads();
    }
    far = simax[0];
    __syncthreads();
  }
  if (tid < NPOINT) {
    int f = sfid[tid];
    #pragma unroll
    for (int c = 0; c < 3; ++c) {
      float v = px[c * NPTS + f];
      nxyz[((size_t)b * NPOINT + tid) * 3 + c] = v;       // (b,s,3) compact
      out[((size_t)b * 3 + c) * NPOINT + tid] = v;        // d_out (B,3,512)
    }
  }
}

// ---------------------------------------------------------------- ball query
__global__ __launch_bounds__(256) void ballquery_kernel(const float* __restrict__ points,
                                                        const float* __restrict__ nxyz,
                                                        int* __restrict__ gidx,
                                                        float r2, int k) {
  const int lane = threadIdx.x & 31;
  const int q = blockIdx.x * 8 + (threadIdx.x >> 5);     // b*512+s
  const int b = q >> 9;
  const float* px = points + (size_t)b * 3 * NPTS;
  const float cx = nxyz[(size_t)q * 3 + 0];
  const float cy = nxyz[(size_t)q * 3 + 1];
  const float cz = nxyz[(size_t)q * 3 + 2];
  const long long base = (long long)q * k;

  int cnt = 0, first = -1;
  for (int nb = 0; nb < NPTS; nb += 32) {
    int n = nb + lane;
    float dx = px[n] - cx, dy = px[NPTS + n] - cy, dz = px[2 * NPTS + n] - cz;
    bool pred = (dx * dx + dy * dy + dz * dz) <= r2;
    unsigned long long mask = __ballot(pred);
    if (mask) {
      if (first < 0) first = nb + (__ffsll(mask) - 1);
      int pos = __popcll(mask & ((1ull << lane) - 1ull));
      if (pred && (cnt + pos) < k) gidx[base + cnt + pos] = n;
      cnt += __popcll(mask);
      if (cnt >= k) break;
    }
  }
  for (int j = cnt + lane; j < k; j += 32) gidx[base + j] = first;
}

// ---------------------------------------------------------------- grouping
__global__ __launch_bounds__(256) void group_kernel(const float* __restrict__ points,
                                                    const float* __restrict__ feature,
                                                    const int* __restrict__ gidx,
                                                    const float* __restrict__ nxyz,
                                                    _Float16* __restrict__ X0, int K) {
  long long p = (long long)blockIdx.x * 256 + threadIdx.x;
  int b = (int)(p / ((long long)NPOINT * K));
  int s = (int)((p / K) % NPOINT);
  int idx = gidx[p];
  const float* px = points + (size_t)b * 3 * NPTS;
  const float* pf = feature + (size_t)b * 3 * NPTS;
  _Float16* xo = X0 + p * 32;
  #pragma unroll
  for (int c = 0; c < 3; ++c) xo[c] = (_Float16)pf[c * NPTS + idx];
  #pragma unroll
  for (int c = 0; c < 3; ++c)
    xo[3 + c] = (_Float16)(px[c * NPTS + idx] - nxyz[((size_t)b * NPOINT + s) * 3 + c]);
  #pragma unroll
  for (int c = 6; c < 32; ++c) xo[c] = (_Float16)0.f;
}

// ---------------------------------------------------------------- BN stats -> scale/shift
__global__ void scaleshift_kernel(const float* __restrict__ stats,
                                  const float* __restrict__ gamma,
                                  const float* __restrict__ beta,
                                  float* __restrict__ ss, int N, float invM) {
  int n = threadIdx.x;
  if (n < N) {
    float mu  = stats[n] * invM;
    float var = stats[128 + n] * invM - mu * mu;
    float sc  = gamma[n] * rsqrtf(var + EPSV);
    ss[n]       = sc;
    ss[128 + n] = beta[n] - mu * sc;
  }
}

// ---------------------------------------------------------------- fused MLP GEMM
// Compile-time <KS,NT,MODE,TF>: fully unrolled, no guards; B-fragments double
// buffered from LDS, next K-step A chunks prefetched from global.
template<int KS, int NT, int MODE, int TF>
__global__ __launch_bounds__(256) void mlp_gemm(const _Float16* __restrict__ X,
                                                const float* __restrict__ w, int wK,
                                                const float* __restrict__ bias,
                                                const float* __restrict__ iss,
                                                const float* __restrict__ oss,
                                                _Float16* __restrict__ Y,
                                                float* __restrict__ stats,
                                                float* __restrict__ out,
                                                int Ksamp, int ch_base) {
  constexpr int N    = NT * 16;
  constexpr int Kpad = KS * 32;
  __shared__ __attribute__((aligned(32))) _Float16 Wlds[Kpad * N];  // B-frag order
  __shared__ float s_scale[Kpad], s_shift[Kpad];
  __shared__ float s_sum[N], s_sq[N];

  const int tid = threadIdx.x;

  // Pre-swizzle weights into exact B-fragment order:
  // element e of lane ln in (ks,nt): k = ks*32 + 8*(ln>>4) + (e<8 ? e : e+8), n = nt*16+(ln&15)
  for (int idx = tid; idx < Kpad * N; idx += 256) {
    int e    = idx & 15;
    int ln   = (idx >> 4) & 31;
    int rest = idx >> 9;                 // ks*NT + nt
    int ks = rest / NT, nt = rest - ks * NT;
    int k = ks * 32 + ((ln >> 4) << 3) + (e < 8 ? e : e + 8);
    int n = nt * 16 + (ln & 15);
    Wlds[idx] = (_Float16)((k < wK) ? w[(size_t)n * wK + k] : 0.f);
  }
  if constexpr (TF) {
    for (int i = tid; i < Kpad; i += 256) { s_scale[i] = iss[i]; s_shift[i] = iss[128 + i]; }
  }
  if constexpr (MODE != MODE_FINAL) {
    for (int i = tid; i < N; i += 256) { s_sum[i] = 0.f; s_sq[i] = 0.f; }
  }
  __syncthreads();

  const int lane = tid & 31;
  const int wid  = tid >> 5;
  const int g    = lane >> 4;
  const int nl   = lane & 15;
  const long long p0 = ((long long)blockIdx.x * 8 + wid) * 16;
  const _Float16* xr = X + (p0 + nl) * (long long)Kpad;

  v8f acc[NT];
  #pragma unroll
  for (int nt = 0; nt < NT; ++nt) {
    float bv = bias[nt * 16 + nl];
    #pragma unroll
    for (int r = 0; r < 8; ++r) acc[nt][r] = bv;
  }

  v8h c0 = *(const v8h*)(xr + 8 * g);                    // K 8g..8g+7
  v8h c1 = *(const v8h*)(xr + 16 + 8 * g);               // K 16+8g..16+8g+7
  #pragma unroll
  for (int ks = 0; ks < KS; ++ks) {
    v8h n0 = c0, n1 = c1;
    if (ks + 1 < KS) {                                   // prefetch next K step
      n0 = *(const v8h*)(xr + (ks + 1) * 32 + 8 * g);
      n1 = *(const v8h*)(xr + (ks + 1) * 32 + 16 + 8 * g);
    }
    v16h a;
    if constexpr (TF) {
      int kb = ks * 32 + 8 * g;
      #pragma unroll
      for (int j = 0; j < 8; ++j) {
        float f = fmaxf(fmaf((float)c0[j], s_scale[kb + j], s_shift[kb + j]), 0.f);
        a[j] = (_Float16)f;
      }
      #pragma unroll
      for (int j = 0; j < 8; ++j) {
        float f = fmaxf(fmaf((float)c1[j], s_scale[kb + 16 + j], s_shift[kb + 16 + j]), 0.f);
        a[8 + j] = (_Float16)f;
      }
    } else {
      #pragma unroll
      for (int j = 0; j < 8; ++j) { a[j] = c0[j]; a[8 + j] = c1[j]; }
    }
    // double-buffered B fragments
    v16h bcur = *(const v16h*)(&Wlds[(ks * NT) * 512 + lane * 16]);
    #pragma unroll
    for (int nt = 0; nt < NT; ++nt) {
      v16h bnext = bcur;
      if (nt + 1 < NT)
        bnext = *(const v16h*)(&Wlds[(ks * NT + nt + 1) * 512 + lane * 16]);
      acc[nt] = __builtin_amdgcn_wmma_f32_16x16x32_f16(false, a, false, bcur,
                                                       (short)0, acc[nt], false, false);
      bcur = bnext;
    }
    c0 = n0; c1 = n1;
  }

  if constexpr (MODE == MODE_FINAL) {
    int b = (int)(p0 / ((long long)NPOINT * Ksamp));
    int s = (int)((p0 / Ksamp) % NPOINT);
    #pragma unroll
    for (int nt = 0; nt < NT; ++nt) {
      int n = nt * 16 + nl;
      float sc = oss[n], sh = oss[128 + n];
      float m = 0.f;                                      // ReLU folded into max-with-0
      #pragma unroll
      for (int r = 0; r < 8; ++r) m = fmaxf(m, fmaf(acc[nt][r], sc, sh));
      m = fmaxf(m, __shfl_xor(m, 16, 32));                // combine rows 0-7 / 8-15
      if (!(lane & 16)) {
        size_t oi = OUT_XYZ + (((size_t)b * TOTCH + ch_base + n) * NPOINT + s);
        atomicMax((unsigned int*)(out + oi), __float_as_uint(m));  // m>=0, out zeroed
      }
    }
  } else {
    #pragma unroll
    for (int nt = 0; nt < NT; ++nt) {
      int n = nt * 16 + nl;
      float sum = 0.f, sq = 0.f;
      #pragma unroll
      for (int r = 0; r < 8; ++r) { float v = acc[nt][r]; sum += v; sq += v * v; }
      atomicAdd(&s_sum[n], sum);
      atomicAdd(&s_sq[n], sq);
      if constexpr (MODE == MODE_STORE) {
        #pragma unroll
        for (int r = 0; r < 8; ++r) {
          long long mrow = p0 + r + 8 * g;
          Y[mrow * N + n] = (_Float16)acc[nt][r];
        }
      }
    }
    __syncthreads();
    if (tid < N) {
      atomicAdd(&stats[tid],       s_sum[tid]);
      atomicAdd(&stats[128 + tid], s_sq[tid]);
    }
  }
}

// ---------------------------------------------------------------- per-branch driver
template<int K, int N1, int N2, int N3>
static void run_branch(void* const* d_in, int br, const float* points, const float* feature,
                       const float* nxyz, int* gidx, float* stats,
                       float* ss0, float* ss1, float* ss2,
                       _Float16* X0, _Float16* X1, _Float16* X2,
                       float* out, float r2, int ch_base, hipStream_t stream) {
  auto leaf = [&](int ly, int lf) -> const float* {      // 0=b,1=beta,2=gamma,3=w
    return (const float*)d_in[3 + (br * 3 + ly) * 4 + lf];
  };
  constexpr long long M = (long long)NBATCH * NPOINT * K;
  constexpr int GB = (int)(M / 128);                     // 8 waves x 16 rows / block
  const float invM = 1.f / (float)M;

  ballquery_kernel<<<512, 256, 0, stream>>>(points, nxyz, gidx, r2, K);
  group_kernel<<<(int)(M / 256), 256, 0, stream>>>(points, feature, gidx, nxyz, X0, K);

  // layer 1: raw grouped input (Kpad=32), store h1 + stats
  zero_kernel<<<1, 256, 0, stream>>>(stats, 256);
  mlp_gemm<1, N1 / 16, MODE_STORE, 0><<<GB, 256, 0, stream>>>(
      X0, leaf(0, 3), 6, leaf(0, 0), nullptr, nullptr, X1, stats, nullptr, K, 0);
  scaleshift_kernel<<<1, 128, 0, stream>>>(stats, leaf(0, 2), leaf(0, 1), ss0, N1, invM);

  // layer 2: BN1+ReLU fused into A-load, store h2 + stats
  zero_kernel<<<1, 256, 0, stream>>>(stats, 256);
  mlp_gemm<N1 / 32, N2 / 16, MODE_STORE, 1><<<GB, 256, 0, stream>>>(
      X1, leaf(1, 3), N1, leaf(1, 0), ss0, nullptr, X2, stats, nullptr, K, 0);
  scaleshift_kernel<<<1, 128, 0, stream>>>(stats, leaf(1, 2), leaf(1, 1), ss1, N2, invM);

  // layer 3 pass A: stats only (h3 never materialized)
  zero_kernel<<<1, 256, 0, stream>>>(stats, 256);
  mlp_gemm<N2 / 32, N3 / 16, MODE_STATS, 1><<<GB, 256, 0, stream>>>(
      X2, leaf(2, 3), N2, leaf(2, 0), ss1, nullptr, nullptr, stats, nullptr, K, 0);
  scaleshift_kernel<<<1, 128, 0, stream>>>(stats, leaf(2, 2), leaf(2, 1), ss2, N3, invM);

  // layer 3 pass B: recompute + BN3 + ReLU + maxpool(K) -> d_out
  mlp_gemm<N2 / 32, N3 / 16, MODE_FINAL, 1><<<GB, 256, 0, stream>>>(
      X2, leaf(2, 3), N2, leaf(2, 0), ss1, ss2, nullptr, nullptr, out, K, ch_base);
}

// ---------------------------------------------------------------- host
extern "C" void kernel_launch(void* const* d_in, const int* in_sizes, int n_in,
                              void* d_out, int out_size, void* d_ws, size_t ws_size,
                              hipStream_t stream) {
  const float* points    = (const float*)d_in[0];
  const float* feature   = (const float*)d_in[1];
  const int*   fps_start = (const int*)d_in[2];

  char* ws = (char*)d_ws;
  float*    nxyz  = (float*)(ws + (1ll << 16));          // 48 KB
  int*      gidx  = (int*)  (ws + (1ll << 20));          // up to 2 MB
  float*    stats = (float*)(ws + (4ll << 20));          // 256 f
  float*    ss0   = (float*)(ws + (4ll << 20) + 4096);
  float*    ss1   = (float*)(ws + (4ll << 20) + 8192);
  float*    ss2   = (float*)(ws + (4ll << 20) + 12288);
  _Float16* X0    = (_Float16*)(ws + (8ll  << 20));      // 32 MB
  _Float16* X1    = (_Float16*)(ws + (40ll << 20));      // 64 MB
  _Float16* X2    = (_Float16*)(ws + (104ll << 20));     // 96 MB  (~200 MB total)
  float* out = (float*)d_out;

  zero_kernel<<<(out_size + 255) / 256, 256, 0, stream>>>(out, out_size);
  fps_kernel<<<NBATCH, 512, 0, stream>>>(points, fps_start, nxyz, out);

  run_branch<16,  32, 32, 64 >(d_in, 0, points, feature, nxyz, gidx, stats,
                               ss0, ss1, ss2, X0, X1, X2, out, 0.01f, 0,   stream);
  run_branch<32,  64, 64, 128>(d_in, 1, points, feature, nxyz, gidx, stats,
                               ss0, ss1, ss2, X0, X1, X2, out, 0.04f, 64,  stream);
  run_branch<128, 64, 96, 128>(d_in, 2, points, feature, nxyz, gidx, stats,
                               ss0, ss1, ss2, X0, X1, X2, out, 0.16f, 192, stream);

  (void)in_sizes; (void)n_in; (void)ws_size;
}
